// Recursion_17635135717960
// MI455X (gfx1250) — matrix operand order repaired and checked by reading, not compiled, not run on hardware
//
#include <hip/hip_runtime.h>
#include <hip/hip_bf16.h>

// ---------------- problem constants (from reference) ----------------
#define T_STEPS 1000
#define DELAY_N 5
#define RING_N  7                      // DELAY+2: write slot never aliases fb slot
#define BATCH   256
#define EDIM    2
#define HID     128
#define DT_C    0.01f
#define DAMP_C  0.5f

// ---------------- tiling ----------------
#define ROWS_PER_BLK 16                // one WMMA M-tile of batch rows per WGP
#define WAVES        8                 // 8 waves x 16 cols = H=128
#define BLK          (WAVES * 32)      // wave32 on gfx1250

typedef __attribute__((ext_vector_type(16))) _Float16 v16h;
typedef __attribute__((ext_vector_type(8)))  float    v8f;

struct B128 { unsigned int w[4]; };
struct B256 { B128 lo, hi; };
struct H16  { _Float16 h[16]; };

// A-fragment (16x32 f16, row-major in LDS) in contiguous k order (B matches):
// lanes 0-15: row M=lane, K 0..15 (VGPR0-7);
// lanes 16-31: row M=lane-16, K 16..31.
static __device__ inline v16h load_a_frag(const _Float16* rowbase, int lane) {
  const int c0 = (lane & 16) ? 16 : 0;
  B256 b;
  b.lo = *(const B128*)(rowbase + c0);        // 16B-aligned ds_load_b128
  b.hi = *(const B128*)(rowbase + c0 + 8);
  return __builtin_bit_cast(v16h, b);
}

// Workgroup barrier that only fences LDS (DScnt), NOT global stores:
// keeps the per-step HBM history writes off the serial critical path.
static __device__ inline void lds_barrier() {
  asm volatile("s_wait_dscnt 0x0\n"
               "s_barrier_signal -1\n"
               "s_barrier_wait -1" ::: "memory");
}

// Branch-free tanh: 1 - 2/(exp2(2*log2e*x)+1); saturates to +/-1 correctly.
static __device__ inline float fast_tanh(float x) {
  float e = __builtin_amdgcn_exp2f(x * 2.88539008177793f);   // exp(2x)
  return 1.0f - 2.0f * __builtin_amdgcn_rcpf(e + 1.0f);
}

__global__ __launch_bounds__(BLK) void rnn_ctrl_kernel(
    const float* __restrict__ gin,  const float* __restrict__ sys0,
    const float* __restrict__ hid0, const float* __restrict__ Win,
    const float* __restrict__ Wrec, const float* __restrict__ bias,
    const float* __restrict__ Wout, float* __restrict__ out_sys,
    float* __restrict__ out_hid)
{
  __shared__ __align__(16) _Float16 hidring[RING_N][ROWS_PER_BLK][HID];  // 28 KB
  __shared__ __align__(16) _Float16 smallA[2][ROWS_PER_BLK][32];         // dbl-buf K-blk 0
  __shared__ float sysring[RING_N][2][ROWS_PER_BLK][EDIM];               // pos/vel ring

  const int tid  = threadIdx.x;
  const int lane = tid & 31;
  const int wv   = tid >> 5;              // N-tile owned by this wave
  const int row0 = blockIdx.x * ROWS_PER_BLK;
  const int colG = wv * 16 + (lane & 15); // hidden column in C/D fragment
  const int mOff = (lane & 16) ? 8 : 0;   // C layout: VGPR v -> M = v (+8 hi lanes)
  const int mrow = lane & 15;

  // ---------------- one-time init ----------------
  for (int idx = tid; idx < ROWS_PER_BLK * HID; idx += BLK) {
    int m = idx / HID, c = idx % HID;
    float h0 = hid0[(row0 + m) * HID + c];
    hidring[0][m][c] = (_Float16)h0;
    out_hid[(size_t)(row0 + m) * HID + c] = h0;      // time slot 0
    #pragma unroll
    for (int s = 1; s < RING_N; ++s) hidring[s][m][c] = (_Float16)0.f;
  }
  for (int idx = tid; idx < 2 * ROWS_PER_BLK * EDIM; idx += BLK) {
    int p = idx / (ROWS_PER_BLK * EDIM);
    int r = idx % (ROWS_PER_BLK * EDIM);
    int m = r / EDIM, d = r % EDIM;
    float v = sys0[(size_t)p * BATCH * EDIM + (row0 + m) * EDIM + d];
    sysring[0][p][m][d] = v;
    out_sys[(size_t)p * BATCH * EDIM + (row0 + m) * EDIM + d] = v;  // slot 0
    #pragma unroll
    for (int s = 1; s < RING_N; ++s) sysring[s][p][m][d] = 0.f;
  }
  for (int idx = tid; idx < 2 * ROWS_PER_BLK * 32; idx += BLK) {
    int b = idx >> 9, m = (idx >> 5) & 15, c = idx & 31;  // [goal, fb_pos=0, fb_vel=0]
    smallA[b][m][c] = (_Float16)((c < EDIM) ? gin[(row0 + m) * EDIM + c] : 0.f);
  }

  // ---------------- static weight B-fragments held in VGPRs ----------------
  // B frag (32x16 f16): lanes 0-15 hold K 0..15 (elem e <-> K=e) at col lane&15,
  // lanes 16-31 hold K 16..31.
  v16h Bf[9];
  v16h BfO[4];                           // zero-padded W_out (H,2) -> 128x16
  {
    const int kHalf = (lane & 16) ? 16 : 0;
    #pragma unroll
    for (int kb = 0; kb < 9; ++kb) {
      H16 s;
      #pragma unroll
      for (int e = 0; e < 16; ++e) {
        int k = kHalf + e;
        float w;
        if (kb == 0)      w = (k < 3 * EDIM) ? Win[k * HID + colG] : 0.f;
        else if (kb <= 4) w = Win[(3 * EDIM + (kb - 1) * 32 + k) * HID + colG];
        else              w = Wrec[((kb - 5) * 32 + k) * HID + colG];
        s.h[e] = (_Float16)w;
      }
      Bf[kb] = __builtin_bit_cast(v16h, s);
    }
    const int colO = lane & 15;
    #pragma unroll
    for (int kb = 0; kb < 4; ++kb) {
      H16 s;
      #pragma unroll
      for (int e = 0; e < 16; ++e) {
        int k = kHalf + e;
        s.h[e] = (_Float16)((colO < EDIM) ? Wout[(kb * 32 + k) * EDIM + colO] : 0.f);
      }
      BfO[kb] = __builtin_bit_cast(v16h, s);
    }
  }
  const float bcol = bias[colG];

  __syncthreads();                       // full fence once (covers global loads)

  // ---------------- 999-step persistent recurrence ----------------
  int cs = 0;                            // slot(t) = t % 7
  #pragma unroll 1
  for (int i = 0; i < T_STEPS - 1; ++i) {
    int ns  = cs + 1; if (ns  >= RING_N) ns  -= RING_N;  // write slot: state(i+1)
    int fbs = cs + 2; if (fbs >= RING_N) fbs -= RING_N;  // holds state(i-5)/zeros
    int fs2 = cs + 3; if (fs2 >= RING_N) fs2 -= RING_N;  // holds state(i-4)/zeros

    // A fragments: no slot aliases the write slot -> no pre-barrier needed
    v16h a0  = load_a_frag(&smallA[i & 1][mrow][0],    lane);
    v16h af0 = load_a_frag(&hidring[fbs][mrow][0],     lane);
    v16h af1 = load_a_frag(&hidring[fbs][mrow][32],    lane);
    v16h af2 = load_a_frag(&hidring[fbs][mrow][64],    lane);
    v16h af3 = load_a_frag(&hidring[fbs][mrow][96],    lane);

    v8f acc;
    #pragma unroll
    for (int v = 0; v < 8; ++v) acc[v] = bcol;
    acc = __builtin_amdgcn_wmma_f32_16x16x32_f16(false, a0,  false, Bf[0], (short)0, acc, false, false);
    acc = __builtin_amdgcn_wmma_f32_16x16x32_f16(false, af0, false, Bf[1], (short)0, acc, false, false);
    acc = __builtin_amdgcn_wmma_f32_16x16x32_f16(false, af1, false, Bf[2], (short)0, acc, false, false);
    acc = __builtin_amdgcn_wmma_f32_16x16x32_f16(false, af2, false, Bf[3], (short)0, acc, false, false);
    acc = __builtin_amdgcn_wmma_f32_16x16x32_f16(false, af3, false, Bf[4], (short)0, acc, false, false);
    #pragma unroll
    for (int kb = 0; kb < 4; ++kb) {     // hid @ W_rec (slot cs is stable this step)
      v16h ah = load_a_frag(&hidring[cs][mrow][kb * 32], lane);
      acc = __builtin_amdgcn_wmma_f32_16x16x32_f16(false, ah, false, Bf[5 + kb], (short)0, acc, false, false);
    }

    // activation + commit new hidden (ring + global history; history drains async)
    float* ohid = out_hid + (size_t)(i + 1) * (BATCH * HID);
    #pragma unroll
    for (int v = 0; v < 8; ++v) {
      int m = mOff + v;
      float h = fast_tanh(acc[v]);
      hidring[ns][m][colG] = (_Float16)h;
      ohid[(row0 + m) * HID + colG] = h;
    }
    lds_barrier();                       // (A) new_hid visible in LDS

    // wave0: force = new_hid @ W_out via 4 WMMAs, Euler, prep next smallA
    if (wv == 0) {
      v8f fa;
      #pragma unroll
      for (int v = 0; v < 8; ++v) fa[v] = 0.f;
      #pragma unroll
      for (int kb = 0; kb < 4; ++kb) {
        v16h ah = load_a_frag(&hidring[ns][mrow][kb * 32], lane);
        fa = __builtin_amdgcn_wmma_f32_16x16x32_f16(false, ah, false, BfO[kb], (short)0, fa, false, false);
      }
      const int d  = lane & 15;          // force lives in cols 0..1
      const int nb = (i + 1) & 1;
      if (d < EDIM) {
        #pragma unroll
        for (int v = 0; v < 8; ++v) {
          int m = mOff + v;
          float f   = fa[v];
          float vel = sysring[cs][1][m][d];
          float pos = sysring[cs][0][m][d];
          float nv  = vel + DT_C * (f - DAMP_C * vel);   // MASS = 1
          float np  = pos + DT_C * nv;
          sysring[ns][0][m][d] = np;
          sysring[ns][1][m][d] = nv;
          size_t ob = (size_t)(i + 1) * (2 * BATCH * EDIM) + (row0 + m) * EDIM + d;
          out_sys[ob] = np;
          out_sys[ob + (size_t)BATCH * EDIM] = nv;
          // step i+1 feedback sys = state(i-4), in slot (i+3)%7
          smallA[nb][m][EDIM + d]     = (_Float16)sysring[fs2][0][m][d];
          smallA[nb][m][2 * EDIM + d] = (_Float16)sysring[fs2][1][m][d];
        }
      }
    }
    lds_barrier();                       // (B) smallA/sys ready for next step

    cs = ns;
  }
}

extern "C" void kernel_launch(void* const* d_in, const int* in_sizes, int n_in,
                              void* d_out, int out_size, void* d_ws, size_t ws_size,
                              hipStream_t stream) {
  (void)in_sizes; (void)n_in; (void)out_size; (void)d_ws; (void)ws_size;
  const float* gin  = (const float*)d_in[0];   // input  (B,D)
  const float* sys0 = (const float*)d_in[1];   // sys0   (2,B,D)
  const float* hid0 = (const float*)d_in[2];   // hidden0(B,H)
  const float* Win  = (const float*)d_in[3];   // W_in   (134,H)
  const float* Wrec = (const float*)d_in[4];   // W_rec  (H,H)
  const float* bvec = (const float*)d_in[5];   // b      (H,)
  const float* Wout = (const float*)d_in[6];   // W_out  (H,D)
  float* out     = (float*)d_out;
  float* out_sys = out;                                        // (T,2,B,D)
  float* out_hid = out + (size_t)T_STEPS * 2 * BATCH * EDIM;   // (T,B,H)

  dim3 grid(BATCH / ROWS_PER_BLK), block(BLK);
  hipLaunchKernelGGL(rnn_ctrl_kernel, grid, block, 0, stream,
                     gin, sys0, hid0, Win, Wrec, bvec, Wout, out_sys, out_hid);
}
